// Bidirectional_LSTM_Encoder_21234318311663
// MI455X (gfx1250) — compile-verified, matrix operand order checked by reading, not run on hardware
//
#include <hip/hip_runtime.h>
#include <hip/hip_bf16.h>

// ---------------------------------------------------------------------------
// Bidirectional 2-layer LSTM encoder for MI455X (gfx1250, wave32, WMMA).
// B=64, T=512, IN=H=512, L=2.  bf16 WMMA (v_wmma_f32_16x16x32_bf16), f32 accum.
// One 1024-thread workgroup (32 waves) per direction:
//   - h ping-pong (2 x 64KB bf16) and x ping-pong (2 x 64KB bf16) in WGP LDS
//   - c state lives entirely in registers (static (m,j) ownership per lane)
//   - x_{t+1} staged via async global->LDS copies overlapped with the GEMM
//   - layer output sequence written via async LDS->global stores
// ---------------------------------------------------------------------------

#define BB   64      // batch
#define TT   512     // time steps
#define KIN  512     // input features
#define HH   512     // hidden
#define NG   2048    // 4*H gate width
#define LL   2       // layers

typedef __attribute__((ext_vector_type(16))) __bf16 bf16x16;
typedef __attribute__((ext_vector_type(8)))  float  f32x8;
typedef __attribute__((ext_vector_type(4)))  int    v4i;

typedef __attribute__((address_space(1))) v4i* g_v4i_p;   // global int4*
typedef __attribute__((address_space(3))) v4i* l_v4i_p;   // LDS int4*

union TileU { bf16x16 v; uint4 q[2]; };

// ---- CDNA5 async global<->LDS copies (ASYNCcnt) ---------------------------
__device__ __forceinline__ void async_g2l_b128(const void* g, const void* l) {
#if __has_builtin(__builtin_amdgcn_global_load_async_to_lds_b128)
  __builtin_amdgcn_global_load_async_to_lds_b128(
      (g_v4i_p)(unsigned long long)g,
      (l_v4i_p)(unsigned)(unsigned long long)l,
      0, 0);
#else
  asm volatile("global_load_async_to_lds_b128 %0, %1, off"
               :: "v"((unsigned)(unsigned long long)l),
                  "v"((unsigned long long)g)
               : "memory");
#endif
}

__device__ __forceinline__ void async_l2g_b128(const void* l, const void* g) {
#if __has_builtin(__builtin_amdgcn_global_store_async_from_lds_b128)
  __builtin_amdgcn_global_store_async_from_lds_b128(
      (g_v4i_p)(unsigned long long)g,
      (l_v4i_p)(unsigned)(unsigned long long)l,
      0, 0);
#else
  asm volatile("global_store_async_from_lds_b128 %0, %1, off"
               :: "v"((unsigned long long)g),
                  "v"((unsigned)(unsigned long long)l)
               : "memory");
#endif
}

__device__ __forceinline__ void wait_async0() {
#if __has_builtin(__builtin_amdgcn_s_wait_asynccnt)
  __builtin_amdgcn_s_wait_asynccnt(0);
#else
  asm volatile("s_wait_asynccnt 0x0" ::: "memory");
#endif
}

// Load a 16x32 bf16 WMMA operand tile (A: rows=M, or B: rows=N since
// W is row-major [n][k] and B[k][n] = W[n][k] -> per-lane k is contiguous).
// CDNA5 16-bit operand layout: lanes 0-15 = row, K 0-7 / 16-23; lanes 16-31 =
// same rows, K 8-15 / 24-31.  Two b128 loads per tile.
__device__ __forceinline__ bf16x16 load_tile16x32(const __bf16* __restrict__ base,
                                                  int row, int row_stride,
                                                  int k0, int lane) {
  const int r  = row + (lane & 15);
  const int kh = (lane >> 4) << 3;           // 0 or 8
  const __bf16* p = base + (size_t)r * row_stride + k0 + kh;
  TileU t;
  t.q[0] = *reinterpret_cast<const uint4*>(p);       // K = k0+kh .. +7
  t.q[1] = *reinterpret_cast<const uint4*>(p + 16);  // K = k0+16+kh .. +7
  return t.v;
}

__device__ __forceinline__ float sigm(float x) {
  return 1.0f / (1.0f + __expf(-x));
}
__device__ __forceinline__ float tanh_fast(float x) {
  return 1.0f - 2.0f / (1.0f + __expf(2.0f * x));
}

#define WMMA_BF16(a, b, c) \
  __builtin_amdgcn_wmma_f32_16x16x32_bf16(false, (a), false, (b), (short)0, (c), false, false)

// ---------------------------------------------------------------------------
// Recurrent scan: grid.x = direction (0 fwd / 1 bwd), 1024 threads = 32 waves.
// K loop = 1024: first 512 from staged x_t (LDS), last 512 from h_{t-1} (LDS).
// ---------------------------------------------------------------------------
__global__ __launch_bounds__(1024, 1)
void lstm_scan_wmma(const __bf16* __restrict__ in_seq_base,  // [dir][T][B][512] bf16
                    const __bf16* __restrict__ wi_base,      // [dir][2048][512] bf16
                    const __bf16* __restrict__ wh_base,      // [dir][2048][512] bf16
                    const float*  __restrict__ bias_base,    // [dir][2048] f32 (b_ih+b_hh)
                    __bf16*       __restrict__ out_seq_base, // [dir][T][B][512] bf16
                    float*        __restrict__ out_final,    // d_out "out" base or null
                    float*        __restrict__ hT_out,       // d_out h[l] base
                    float*        __restrict__ cT_out,       // d_out c[l] base
                    int is_last) {
  const int dir  = blockIdx.x;
  const int tid  = threadIdx.x;
  const int w    = tid >> 5;          // wave 0..31
  const int lane = tid & 31;
  const int mt   = w & 3;             // M tile (4 x 16 = 64 batch rows)
  const int hs   = w >> 2;            // hidden slice 0..7 (4 hidden tiles each)
  const int dcol = dir * HH;

  const __bf16* in_seq = in_seq_base + (size_t)dir * TT * BB * KIN;
  const __bf16* wi     = wi_base     + (size_t)dir * NG * KIN;
  const __bf16* wh     = wh_base     + (size_t)dir * NG * HH;
  const float*  bias   = bias_base   + (size_t)dir * NG;
  __bf16*       oseq   = out_seq_base + (size_t)dir * TT * BB * HH;

  extern __shared__ char smem[];
  __bf16* hbuf0 = reinterpret_cast<__bf16*>(smem);            //  64 KB
  __bf16* hbuf1 = reinterpret_cast<__bf16*>(smem + 65536);    //  64 KB
  __bf16* xbuf0 = reinterpret_cast<__bf16*>(smem + 131072);   //  64 KB
  __bf16* xbuf1 = reinterpret_cast<__bf16*>(smem + 196608);   //  64 KB

  // cell state: statically owned per lane -> registers for the whole scan
  float creg[4][8];
  #pragma unroll
  for (int q = 0; q < 4; ++q)
    #pragma unroll
    for (int r = 0; r < 8; ++r) creg[q][r] = 0.0f;

  // zero h0; stage x_0 into xbuf0
  for (int i = tid; i < BB * HH; i += 1024) hbuf0[i] = (__bf16)0.0f;
  #pragma unroll
  for (int ii = 0; ii < 4; ++ii) {
    const int ch = tid + ii * 1024;               // 16B chunks: 4096 total
    async_g2l_b128(in_seq + (size_t)ch * 8, xbuf0 + (size_t)ch * 8);
  }
  wait_async0();
  __syncthreads();

  for (int t = 0; t < TT; ++t) {
    const __bf16* xcur = (t & 1) ? xbuf1 : xbuf0;
    __bf16*       xnxt = (t & 1) ? xbuf0 : xbuf1;
    const __bf16* hcur = (t & 1) ? hbuf1 : hbuf0;
    __bf16*       hnxt = (t & 1) ? hbuf0 : hbuf1;

    // async: stage x_{t+1} (overlaps with this step's GEMM)
    if (t + 1 < TT) {
      const __bf16* src = in_seq + (size_t)(t + 1) * BB * KIN;
      #pragma unroll
      for (int ii = 0; ii < 4; ++ii) {
        const int ch = tid + ii * 1024;
        async_g2l_b128(src + (size_t)ch * 8, xnxt + (size_t)ch * 8);
      }
    }
    // async: drain h_{t-1} (already complete in LDS) to the layer output seq
    if (!is_last && t > 0) {
      __bf16* dst = oseq + (size_t)(t - 1) * BB * HH;
      #pragma unroll
      for (int ii = 0; ii < 4; ++ii) {
        const int ch = tid + ii * 1024;
        async_l2g_b128(hcur + (size_t)ch * 8, dst + (size_t)ch * 8);
      }
    }

    #pragma unroll
    for (int qi = 0; qi < 4; ++qi) {
      const int hjt  = hs * 4 + qi;            // hidden tile 0..31
      const int jcol = hjt * 16 + (lane & 15); // this lane's hidden column

      const __bf16* wi0 = wi + (size_t)(0 * HH + hjt * 16) * KIN;
      const __bf16* wi1 = wi + (size_t)(1 * HH + hjt * 16) * KIN;
      const __bf16* wi2 = wi + (size_t)(2 * HH + hjt * 16) * KIN;
      const __bf16* wi3 = wi + (size_t)(3 * HH + hjt * 16) * KIN;
      const __bf16* wh0 = wh + (size_t)(0 * HH + hjt * 16) * HH;
      const __bf16* wh1 = wh + (size_t)(1 * HH + hjt * 16) * HH;
      const __bf16* wh2 = wh + (size_t)(2 * HH + hjt * 16) * HH;
      const __bf16* wh3 = wh + (size_t)(3 * HH + hjt * 16) * HH;

      // init accumulators with fused bias (same column for all 8 rows)
      f32x8 acc0, acc1, acc2, acc3;
      {
        const float b0 = bias[0 * HH + jcol], b1 = bias[1 * HH + jcol];
        const float b2 = bias[2 * HH + jcol], b3 = bias[3 * HH + jcol];
        #pragma unroll
        for (int r = 0; r < 8; ++r) { acc0[r] = b0; acc1[r] = b1; acc2[r] = b2; acc3[r] = b3; }
      }

      // ---- input half of K (x_t @ Wi^T): A from LDS, B tiles hoisted ----
      #pragma unroll 2
      for (int kk = 0; kk < 16; ++kk) {
        const bf16x16 a  = load_tile16x32(xcur, mt * 16, KIN, kk * 32, lane);
        const bf16x16 b0 = load_tile16x32(wi0, 0, KIN, kk * 32, lane);
        const bf16x16 b1 = load_tile16x32(wi1, 0, KIN, kk * 32, lane);
        const bf16x16 b2 = load_tile16x32(wi2, 0, KIN, kk * 32, lane);
        const bf16x16 b3 = load_tile16x32(wi3, 0, KIN, kk * 32, lane);
        acc0 = WMMA_BF16(a, b0, acc0);
        acc1 = WMMA_BF16(a, b1, acc1);
        acc2 = WMMA_BF16(a, b2, acc2);
        acc3 = WMMA_BF16(a, b3, acc3);
      }
      // ---- recurrent half of K (h_{t-1} @ Wh^T): A from LDS ----
      #pragma unroll 2
      for (int kk = 0; kk < 16; ++kk) {
        const bf16x16 a  = load_tile16x32(hcur, mt * 16, HH, kk * 32, lane);
        const bf16x16 b0 = load_tile16x32(wh0, 0, HH, kk * 32, lane);
        const bf16x16 b1 = load_tile16x32(wh1, 0, HH, kk * 32, lane);
        const bf16x16 b2 = load_tile16x32(wh2, 0, HH, kk * 32, lane);
        const bf16x16 b3 = load_tile16x32(wh3, 0, HH, kk * 32, lane);
        acc0 = WMMA_BF16(a, b0, acc0);
        acc1 = WMMA_BF16(a, b1, acc1);
        acc2 = WMMA_BF16(a, b2, acc2);
        acc3 = WMMA_BF16(a, b3, acc3);
      }

      // ---- in-register cell update (this wave owns all 4 gates of (m,j)) ----
      #pragma unroll
      for (int r = 0; r < 8; ++r) {
        const int m = mt * 16 + r + ((lane >> 4) << 3);  // C/D layout row
        const int j = jcol;
        const float iv = sigm(acc0[r]);
        const float fv = sigm(acc1[r]);
        const float gv = tanh_fast(acc2[r]);
        const float ov = sigm(acc3[r]);
        const float cn = fv * creg[qi][r] + iv * gv;
        const float hv = ov * tanh_fast(cn);
        creg[qi][r] = cn;
        hnxt[m * HH + j] = (__bf16)hv;
        if (is_last) {
          const int t_orig = dir ? (TT - 1 - t) : t;   // un-reverse bwd stream
          out_final[((size_t)m * TT + t_orig) * (2 * HH) + dcol + j] = hv;
        }
        if (t == TT - 1) {
          hT_out[m * (2 * HH) + dcol + j] = hv;
          cT_out[m * (2 * HH) + dcol + j] = hv;   // reference returns c = h
        }
      }
    }

    wait_async0();     // staged x_{t+1} landed; h_{t-1} drained before reuse
    __syncthreads();   // h ping-pong ready for next step
  }

  // drain the final h_{T-1} to the layer output sequence
  if (!is_last) {
    const __bf16* hlast = hbuf0;                 // TT even -> hbuf[(TT)&1]
    __bf16* dst = oseq + (size_t)(TT - 1) * BB * HH;
    #pragma unroll
    for (int ii = 0; ii < 4; ++ii) {
      const int ch = tid + ii * 1024;
      async_l2g_b128(hlast + (size_t)ch * 8, dst + (size_t)ch * 8);
    }
    wait_async0();
  }
}

// ---------------------------------------------------------------------------
// One-time conversions: x -> time-major bf16 (fwd + time-reversed bwd stream),
// weights -> bf16 [l][dir][n][k], bias -> b_ih + b_hh (f32).
// ---------------------------------------------------------------------------
__global__ void conv_x_kernel(const float* __restrict__ x, __bf16* __restrict__ xtm) {
  const size_t total = (size_t)2 * TT * BB * KIN;   // 2^25
  for (size_t i = blockIdx.x * blockDim.x + threadIdx.x; i < total;
       i += (size_t)gridDim.x * blockDim.x) {
    const int k   = (int)(i & 511);
    const int m   = (int)((i >> 9) & 63);
    const int t   = (int)((i >> 15) & 511);
    const int dir = (int)(i >> 24);
    const int ts  = dir ? (TT - 1 - t) : t;
    xtm[i] = (__bf16)x[((size_t)m * TT + ts) * KIN + k];
  }
}

__global__ void conv_w_kernel(const float* __restrict__ wi_f, const float* __restrict__ wi_b,
                              const float* __restrict__ wh_f, const float* __restrict__ wh_b,
                              __bf16* __restrict__ wi_o, __bf16* __restrict__ wh_o) {
  const size_t total = (size_t)LL * 2 * NG * KIN;   // 2^22
  for (size_t i = blockIdx.x * blockDim.x + threadIdx.x; i < total;
       i += (size_t)gridDim.x * blockDim.x) {
    const int k   = (int)(i & 511);
    const int n   = (int)((i >> 9) & 2047);
    const int dir = (int)((i >> 20) & 1);
    const int l   = (int)(i >> 21);
    const size_t src = ((size_t)l * NG + n) * KIN + k;
    wi_o[i] = (__bf16)(dir ? wi_b[src] : wi_f[src]);
    wh_o[i] = (__bf16)(dir ? wh_b[src] : wh_f[src]);
  }
}

__global__ void conv_b_kernel(const float* __restrict__ bi_f, const float* __restrict__ bh_f,
                              const float* __restrict__ bi_b, const float* __restrict__ bh_b,
                              float* __restrict__ bias_o) {
  const size_t total = (size_t)LL * 2 * NG;         // 8192
  for (size_t i = blockIdx.x * blockDim.x + threadIdx.x; i < total;
       i += (size_t)gridDim.x * blockDim.x) {
    const int n   = (int)(i & 2047);
    const int dir = (int)((i >> 11) & 1);
    const int l   = (int)(i >> 12);
    const size_t src = (size_t)l * NG + n;
    bias_o[i] = dir ? (bi_b[src] + bh_b[src]) : (bi_f[src] + bh_f[src]);
  }
}

// ---------------------------------------------------------------------------
extern "C" void kernel_launch(void* const* d_in, const int* in_sizes, int n_in,
                              void* d_out, int out_size, void* d_ws, size_t ws_size,
                              hipStream_t stream) {
  (void)in_sizes; (void)n_in; (void)out_size; (void)ws_size;
  const float* x     = (const float*)d_in[0];
  const float* wi_f  = (const float*)d_in[1];
  const float* wh_f  = (const float*)d_in[2];
  const float* bi_f  = (const float*)d_in[3];
  const float* bh_f  = (const float*)d_in[4];
  const float* wi_b  = (const float*)d_in[5];
  const float* wh_b  = (const float*)d_in[6];
  const float* bi_b  = (const float*)d_in[7];
  const float* bh_b  = (const float*)d_in[8];
  float* outF = (float*)d_out;

  // d_out layout: out (64,512,1024) | h (2,64,1024) | c (2,64,1024)
  const size_t OUT_ELEMS = (size_t)BB * TT * 2 * HH;   // 33,554,432
  const size_t H_OFF     = OUT_ELEMS;
  const size_t C_OFF     = OUT_ELEMS + (size_t)LL * BB * 2 * HH;
  const size_t H_LSTRIDE = (size_t)BB * 2 * HH;        // 65,536 per layer

  // workspace carve-up (bf16 = 2B)
  char* ws = (char*)d_ws;
  size_t off = 0;
  __bf16* xtm  = (__bf16*)(ws + off); off += (size_t)2 * TT * BB * KIN * 2;  // 64 MB
  __bf16* seqA = (__bf16*)(ws + off); off += (size_t)2 * TT * BB * HH  * 2;  // 64 MB
  __bf16* wi   = (__bf16*)(ws + off); off += (size_t)LL * 2 * NG * KIN * 2;  //  8 MB
  __bf16* wh   = (__bf16*)(ws + off); off += (size_t)LL * 2 * NG * HH  * 2;  //  8 MB
  float*  bias = (float*) (ws + off); off += (size_t)LL * 2 * NG * 4;        // 32 KB

  conv_x_kernel<<<2048, 256, 0, stream>>>(x, xtm);
  conv_w_kernel<<<2048, 256, 0, stream>>>(wi_f, wi_b, wh_f, wh_b, wi, wh);
  conv_b_kernel<<<32, 256, 0, stream>>>(bi_f, bh_f, bi_b, bh_b, bias);

  const size_t WSTRIDE = (size_t)2 * NG * KIN;   // per-layer weight stride (both dirs)
  const size_t BSTRIDE = (size_t)2 * NG;
  const size_t SHMEM   = 262144;                 // h0+h1+x0+x1 = 256 KB (<= 320 KB WGP LDS)

  // layer 0: x -> seqA  (grid = 2 directions, each on its own WGP)
  lstm_scan_wmma<<<2, 1024, SHMEM, stream>>>(
      xtm, wi, wh, bias, seqA,
      nullptr, outF + H_OFF, outF + C_OFF, /*is_last=*/0);

  // layer 1: seqA -> d_out (out section), seq scratch unused
  lstm_scan_wmma<<<2, 1024, SHMEM, stream>>>(
      seqA, wi + WSTRIDE, wh + WSTRIDE, bias + BSTRIDE, xtm,
      outF, outF + H_OFF + H_LSTRIDE, outF + C_OFF + H_LSTRIDE, /*is_last=*/1);
}
